// NLNN_82892868813187
// MI455X (gfx1250) — matrix operand order repaired and checked by reading, not compiled, not run on hardware
//
#include <hip/hip_runtime.h>
#include <hip/hip_bf16.h>

// ---------------------------------------------------------------------------
// Non-local block (theta/phi/g 1x1 convs -> dense softmax attention -> 1x1
// conv + residual) for MI455X (gfx1250, wave32, WMMA bf16 16x16x32).
//
// All GEMMs run on v_wmma_f32_16x16x32_bf16 with fp32 accumulation.
// The reference's raw .view() reshapes are pure reinterpretations, handled
// here as stride tricks on the flat projection buffers.
//
// Workspace layout (requires ws_size >= 75,497,472 bytes):
//   proj bf16 [B][1536][2304]  (rows 0-511 theta, 512-1023 phi, 1024-1535 g)
//   ybuf bf16 [B][2304][512]   (attention output, pre-final-conv)
// ---------------------------------------------------------------------------

#define CIN   1024
#define CMID  512
#define NB    8
#define HW    2304
#define PROJR 1536

typedef __attribute__((ext_vector_type(16))) __bf16 bf16x16;
typedef __attribute__((ext_vector_type(8)))  __bf16 bf16x8;
typedef __attribute__((ext_vector_type(8)))  float  f32x8;
typedef __attribute__((ext_vector_type(4)))  float  f32x4;

#define WMMA_BF16(a, b, c) \
  __builtin_amdgcn_wmma_f32_16x16x32_bf16(false, (a), false, (b), (short)0, (c), false, false)

// A fragment (16x32, M x K), rows [m0..m0+15], LDS tile row-major, row stride
// 'stride' (bf16 elems, multiple of 8). Lane half 0: K = k0+{0..7, 16..23};
// half 1: K = k0+{8..15, 24..31}.
__device__ __forceinline__ bf16x16 frag_a(const __bf16* base, int stride,
                                          int m0, int k0, int lane) {
  const __bf16* p = base + (m0 + (lane & 15)) * stride + k0 + ((lane >> 4) << 3);
  bf16x8 lo = *(const bf16x8*)(p);
  bf16x8 hi = *(const bf16x8*)(p + 16);
  return __builtin_shufflevector(lo, hi, 0, 1, 2, 3, 4, 5, 6, 7,
                                 8, 9, 10, 11, 12, 13, 14, 15);
}

// B fragment (32x16, K x N), cols [n0..n0+15], LDS tile col-major (k
// contiguous per column), col stride 'stride'. Lane half 0: K = k0+{0..7, 16..23};
// half 1: K = k0+{8..15, 24..31}.
__device__ __forceinline__ bf16x16 frag_b(const __bf16* base, int stride,
                                          int k0, int n0, int lane) {
  const __bf16* p = base + (n0 + (lane & 15)) * stride + k0 + ((lane >> 4) << 3);
  bf16x8 lo = *(const bf16x8*)(p);
  bf16x8 hi = *(const bf16x8*)(p + 16);
  return __builtin_shufflevector(lo, hi, 0, 1, 2, 3, 4, 5, 6, 7,
                                 8, 9, 10, 11, 12, 13, 14, 15);
}

// ---------------------------------------------------------------------------
// Kernel 1: proj[b] = [w1;w2;w3] (1536x1024) @ x[b] (1024x2304), bf16 out.
// Grid (12, 36, 8), block 256 (8 waves); workgroup tile 128x64, wave 32x32.
// ---------------------------------------------------------------------------
__global__ __launch_bounds__(256) void proj_gemm_kernel(
    const float* __restrict__ x,
    const float* __restrict__ w1, const float* __restrict__ b1,
    const float* __restrict__ w2, const float* __restrict__ b2,
    const float* __restrict__ w3, const float* __restrict__ b3,
    __bf16* __restrict__ proj) {
  __shared__ __bf16 At[128 * 40];  // [row][k], padded stride 40
  __shared__ __bf16 Bt[64 * 40];   // [col][k] (transposed), padded stride 40

  const int tid = threadIdx.x, lane = tid & 31, wid = tid >> 5;
  const int m0 = blockIdx.x * 128;
  const int n0 = blockIdx.y * 64;
  const int b  = blockIdx.z;

  const float* w;
  const float* bias;
  int mbase;
  if (m0 < 512)        { w = w1; bias = b1; mbase = m0; }
  else if (m0 < 1024)  { w = w2; bias = b2; mbase = m0 - 512; }
  else                 { w = w3; bias = b3; mbase = m0 - 1024; }

  const float* xb = x + (size_t)b * CIN * HW;
  const int wr = (wid >> 1) * 32;  // wave row offset in tile
  const int wc = (wid & 1) * 32;   // wave col offset in tile

  f32x8 acc[2][2] = {};

  for (int k0 = 0; k0 < CIN; k0 += 32) {
    __syncthreads();
    // A: 128 rows x 32 k, fp32 -> bf16
#pragma unroll
    for (int i = 0; i < 4; ++i) {
      int e = tid + i * 256;         // 0..1023 float4s
      int row = e >> 3;
      int k4  = (e & 7) << 2;
      f32x4 v = *(const f32x4*)(w + (size_t)(mbase + row) * CIN + k0 + k4);
#pragma unroll
      for (int j = 0; j < 4; ++j) At[row * 40 + k4 + j] = (__bf16)v[j];
    }
    // B: 32 k x 64 n, fp32 -> bf16, transposed to [n][k]
#pragma unroll
    for (int i = 0; i < 2; ++i) {
      int e = tid + i * 256;         // 0..511 float4s
      int kr = e >> 4;
      int c4 = (e & 15) << 2;
      f32x4 v = *(const f32x4*)(xb + (size_t)(k0 + kr) * HW + n0 + c4);
#pragma unroll
      for (int j = 0; j < 4; ++j) Bt[(c4 + j) * 40 + kr] = (__bf16)v[j];
    }
    __syncthreads();
#pragma unroll
    for (int mi = 0; mi < 2; ++mi) {
      bf16x16 a = frag_a(At, 40, wr + mi * 16, 0, lane);
#pragma unroll
      for (int ni = 0; ni < 2; ++ni) {
        bf16x16 bb = frag_b(Bt, 40, 0, wc + ni * 16, lane);
        acc[mi][ni] = WMMA_BF16(a, bb, acc[mi][ni]);
      }
    }
  }

  // Epilogue: + bias, cast bf16, store to proj[b] as row-major [1536][2304]
  __bf16* ob = proj + (size_t)b * PROJR * HW;
  const int half = lane >> 4, col = lane & 15;
#pragma unroll
  for (int mi = 0; mi < 2; ++mi)
#pragma unroll
    for (int ni = 0; ni < 2; ++ni)
#pragma unroll
      for (int r = 0; r < 8; ++r) {
        int lrow = wr + mi * 16 + 8 * half + r;
        float v  = acc[mi][ni][r] + bias[mbase + lrow];
        ob[(size_t)(m0 + lrow) * HW + n0 + wc + ni * 16 + col] = (__bf16)v;
      }
}

// ---------------------------------------------------------------------------
// Kernel 2: flash-style attention per 32-row tile.
//   theta = proj[b] viewed [2304][512] (raw view of [512][2304] conv output)
//   phi   = proj[b]+512*2304 viewed [512][2304]
//   g     = proj[b]+1024*2304 viewed [2304][512]
//   Y[n,c] = sum_m exp(S[n,m]) g[m,c] / sum_m exp(S[n,m]),  S = theta @ phi.
// No max-subtraction (|S| <~ 50, exp fits fp32/bf16 range); row-sums via an
// extra WMMA with an all-ones B fragment (same C-layout as Y fragments).
// Grid (72, 8), block 256 (8 waves): phase A wave = 16x16 S frag,
// phase B wave = 16 rows x 128 cols of Y (8 accumulators).
// ---------------------------------------------------------------------------
#define ATTN_SMEM 120960

__global__ __launch_bounds__(256) void attn_kernel(
    const __bf16* __restrict__ proj, __bf16* __restrict__ ybuf) {
  extern __shared__ char smem[];
  __bf16* Th = (__bf16*)(smem);            // theta tile [32][520]
  __bf16* Ph = (__bf16*)(smem + 33280);    // phi chunk  [64 m][72 k]
  __bf16* Pt = (__bf16*)(smem + 42496);    // exp(S)     [32][72]
  __bf16* Gt = (__bf16*)(smem + 47104);    // g tile     [512 c][72 m]
  float*  lb = (float*)(smem + 120832);    // row sums   [32]

  const int tid = threadIdx.x, lane = tid & 31, wid = tid >> 5;
  const int n0 = blockIdx.x * 32;
  const int b  = blockIdx.y;

  const __bf16* theta = proj + (size_t)b * PROJR * HW;   // view [2304][512]
  const __bf16* phi   = theta + (size_t)CMID * HW;       // view [512][2304]
  const __bf16* g     = theta + (size_t)2 * CMID * HW;   // view [2304][512]

  // Stage theta rows [n0..n0+31] x 512 once (row-major, k-contiguous).
#pragma unroll
  for (int i = 0; i < 8; ++i) {
    int e = tid + i * 256;         // 0..2047 bf16x8s
    int row = e >> 6;
    int k8  = (e & 63) << 3;
    *(bf16x8*)(Th + row * 520 + k8) =
        *(const bf16x8*)(theta + (size_t)(n0 + row) * CMID + k8);
  }

  bf16x16 onesv;
#pragma unroll
  for (int i = 0; i < 16; ++i) onesv[i] = (__bf16)1.0f;

  const int rg = wid >> 2;          // row group (0..1) -> rows rg*16
  const int cg = wid & 3;           // col group (0..3)
  const int rbase = rg * 16;
  const int half = lane >> 4, col = lane & 15;

  f32x8 yacc[8] = {};
  f32x8 lacc = {};

  for (int nt = 0; nt < 36; ++nt) {
    const int mc = nt * 64;         // attention-column (m) tile base
    f32x8 sacc = {};

    // ---- Phase A: S(32x64) = theta(32x512) @ phi(512x64), k-chunks of 64.
    for (int kc = 0; kc < CMID; kc += 64) {
      __syncthreads();
#pragma unroll
      for (int i = 0; i < 2; ++i) {
        int e = tid + i * 256;       // 0..511 bf16x8s
        int kr = e >> 3;
        int m8 = (e & 7) << 3;
        bf16x8 v = *(const bf16x8*)(phi + (size_t)(kc + kr) * HW + mc + m8);
#pragma unroll
        for (int j = 0; j < 8; ++j) Ph[(m8 + j) * 72 + kr] = v[j];  // transpose
      }
      __syncthreads();
#pragma unroll
      for (int kk = 0; kk < 64; kk += 32) {
        bf16x16 a  = frag_a(Th, 520, rbase, kc + kk, lane);
        bf16x16 bb = frag_b(Ph, 72, kk, cg * 16, lane);
        sacc = WMMA_BF16(a, bb, sacc);
      }
    }

    // ---- exp(S) -> Pt  (wave owns rows rbase..+15, cols cg*16..+15)
#pragma unroll
    for (int r = 0; r < 8; ++r) {
      float p = __expf(sacc[r]);
      Pt[(rbase + 8 * half + r) * 72 + cg * 16 + col] = (__bf16)p;
    }

    // ---- Stage G tile [mc..mc+63] x 512, transposed to [c][m].
#pragma unroll
    for (int i = 0; i < 16; ++i) {
      int e = tid + i * 256;         // 0..4095 bf16x8s
      int m  = e >> 6;
      int c8 = (e & 63) << 3;
      bf16x8 v = *(const bf16x8*)(g + (size_t)(mc + m) * CMID + c8);
#pragma unroll
      for (int j = 0; j < 8; ++j) Gt[(c8 + j) * 72 + m] = v[j];
    }
    __syncthreads();

    // ---- Phase B: Y(32x512) += P(32x64) @ G(64x512); l += P @ ones.
#pragma unroll
    for (int kk = 0; kk < 64; kk += 32) {
      bf16x16 a = frag_a(Pt, 72, rbase, kk, lane);
#pragma unroll
      for (int ci = 0; ci < 8; ++ci) {
        bf16x16 bb = frag_b(Gt, 72, kk, cg * 128 + ci * 16, lane);
        yacc[ci] = WMMA_BF16(a, bb, yacc[ci]);
      }
      if (cg == 0) lacc = WMMA_BF16(a, onesv, lacc);
    }
  }

  // Broadcast row-sums (all 16 cols of lacc hold the row sum).
  if (cg == 0 && col == 0) {
#pragma unroll
    for (int r = 0; r < 8; ++r) lb[rbase + 8 * half + r] = lacc[r];
  }
  __syncthreads();

  // Normalize and store Y as [2304][512] row-major bf16.
  __bf16* yb = ybuf + (size_t)b * HW * CMID;
#pragma unroll
  for (int ci = 0; ci < 8; ++ci)
#pragma unroll
    for (int r = 0; r < 8; ++r) {
      int row = rbase + 8 * half + r;
      float v = yacc[ci][r] / lb[row];
      yb[(size_t)(n0 + row) * CMID + cg * 128 + ci * 16 + col] = (__bf16)v;
    }
}

// ---------------------------------------------------------------------------
// Kernel 3: out[b] = x[b] + b4 + w4(1024x512) @ y_chw(512x2304), where
// y_chw is ybuf[b] reinterpreted [512][2304]. Same tiling as kernel 1.
// Grid (8, 36, 8), block 256.
// ---------------------------------------------------------------------------
__global__ __launch_bounds__(256) void out_gemm_kernel(
    const float* __restrict__ x, const float* __restrict__ w4,
    const float* __restrict__ b4, const __bf16* __restrict__ ybuf,
    float* __restrict__ out) {
  __shared__ __bf16 At[128 * 40];
  __shared__ __bf16 Bt[64 * 40];

  const int tid = threadIdx.x, lane = tid & 31, wid = tid >> 5;
  const int m0 = blockIdx.x * 128;
  const int n0 = blockIdx.y * 64;
  const int b  = blockIdx.z;

  const __bf16* yb = ybuf + (size_t)b * HW * CMID;  // view [512][2304]
  const int wr = (wid >> 1) * 32;
  const int wc = (wid & 1) * 32;

  f32x8 acc[2][2] = {};

  for (int k0 = 0; k0 < CMID; k0 += 32) {
    __syncthreads();
#pragma unroll
    for (int i = 0; i < 4; ++i) {
      int e = tid + i * 256;
      int row = e >> 3;
      int k4  = (e & 7) << 2;
      f32x4 v = *(const f32x4*)(w4 + (size_t)(m0 + row) * CMID + k0 + k4);
#pragma unroll
      for (int j = 0; j < 4; ++j) At[row * 40 + k4 + j] = (__bf16)v[j];
    }
    {
      int kr = tid >> 3;             // 0..31
      int c8 = (tid & 7) << 3;       // 0..56
      bf16x8 v = *(const bf16x8*)(yb + (size_t)(k0 + kr) * HW + n0 + c8);
#pragma unroll
      for (int j = 0; j < 8; ++j) Bt[(c8 + j) * 40 + kr] = v[j];
    }
    __syncthreads();
#pragma unroll
    for (int mi = 0; mi < 2; ++mi) {
      bf16x16 a = frag_a(At, 40, wr + mi * 16, 0, lane);
#pragma unroll
      for (int ni = 0; ni < 2; ++ni) {
        bf16x16 bb = frag_b(Bt, 40, 0, wc + ni * 16, lane);
        acc[mi][ni] = WMMA_BF16(a, bb, acc[mi][ni]);
      }
    }
  }

  const float* xb = x + (size_t)b * CIN * HW;
  float* ob = out + (size_t)b * CIN * HW;
  const int half = lane >> 4, col = lane & 15;
#pragma unroll
  for (int mi = 0; mi < 2; ++mi)
#pragma unroll
    for (int ni = 0; ni < 2; ++ni)
#pragma unroll
      for (int r = 0; r < 8; ++r) {
        int gm = m0 + wr + mi * 16 + 8 * half + r;
        int gp = n0 + wc + ni * 16 + col;
        ob[(size_t)gm * HW + gp] =
            acc[mi][ni][r] + b4[gm] + xb[(size_t)gm * HW + gp];
      }
}

// ---------------------------------------------------------------------------
extern "C" void kernel_launch(void* const* d_in, const int* in_sizes, int n_in,
                              void* d_out, int out_size, void* d_ws,
                              size_t ws_size, hipStream_t stream) {
  (void)in_sizes; (void)n_in; (void)out_size; (void)ws_size;

  const float* x  = (const float*)d_in[0];
  const float* w1 = (const float*)d_in[1];
  const float* b1 = (const float*)d_in[2];
  const float* w2 = (const float*)d_in[3];
  const float* b2 = (const float*)d_in[4];
  const float* w3 = (const float*)d_in[5];
  const float* b3 = (const float*)d_in[6];
  const float* w4 = (const float*)d_in[7];
  const float* b4 = (const float*)d_in[8];
  float* out = (float*)d_out;

  __bf16* proj = (__bf16*)d_ws;                           // [8][1536][2304]
  __bf16* ybuf = proj + (size_t)NB * PROJR * HW;          // [8][2304][512]

  proj_gemm_kernel<<<dim3(12, 36, NB), 256, 0, stream>>>(
      x, w1, b1, w2, b2, w3, b3, proj);
  attn_kernel<<<dim3(HW / 32, NB), 256, ATTN_SMEM, stream>>>(proj, ybuf);
  out_gemm_kernel<<<dim3(8, 36, NB), 256, 0, stream>>>(x, w4, b4, ybuf, out);
}
